// PoPESelfAttention_20091857011451
// MI455X (gfx1250) — matrix operand order repaired and checked by reading, not compiled, not run on hardware
//
#include <hip/hip_runtime.h>
#include <hip/hip_bf16.h>

typedef __attribute__((ext_vector_type(16))) _Float16 v16h;
typedef __attribute__((ext_vector_type(8)))  _Float16 v8h;
typedef __attribute__((ext_vector_type(8)))  float    v8f;
typedef __attribute__((ext_vector_type(4)))  unsigned int u32x4;
typedef __attribute__((ext_vector_type(8)))  int      i32x8;
typedef __attribute__((ext_vector_type(4)))  int      i32x4;

#define D_MODEL  2048
#define N_HEADS  16
#define N_KV     8
#define HD       128
#define SEQ      2048
#define BATCH    2
#define BT       (BATCH * SEQ)        // 4096 rows for projection GEMMs
#define QKDIM    256                  // concatenated real+imag head dim

__device__ __forceinline__ v8f zero8() {
  v8f z;
#pragma unroll
  for (int i = 0; i < 8; ++i) z[i] = 0.0f;
  return z;
}

__device__ __forceinline__ float softplusf(float x) {
  return (x > 15.0f) ? x : log1pf(__expf(x));
}

// ---- WMMA fragment loaders (CDNA5 wave32 layouts, CDNA5 ISA) ----
// A (16x32 f16): lane m=l&15, g=l>>4; elems 0..7 -> K=8g+e ; 8..15 -> K=16+8g+(e-8)
__device__ __forceinline__ v16h load_a_frag(const _Float16* __restrict__ A,
                                            int lda, int lane) {
  const int m = lane & 15, g = lane >> 4;
  const _Float16* p = A + (size_t)m * lda + 8 * g;
  union { v16h v; v8h h[2]; } u;
  u.h[0] = *(const v8h*)(p);
  u.h[1] = *(const v8h*)(p + 16);
  return u.v;
}
// B (32x16 f16) from row-major Bt[N][K]: lane n=l&15, g=l>>4; elems 0..7 -> K=8g+e ; 8..15 -> K=16+8g+(e-8)
__device__ __forceinline__ v16h load_b_frag(const _Float16* __restrict__ Bt,
                                            int ldb, int lane) {
  const int n = lane & 15, g = lane >> 4;
  const _Float16* p = Bt + (size_t)n * ldb + 8 * g;
  union { v16h v; v8h h[2]; } u;
  u.h[0] = *(const v8h*)(p);
  u.h[1] = *(const v8h*)(p + 16);
  return u.v;
}

__device__ __forceinline__ v8f wmma16(v16h a, v16h b, v8f c) {
  return __builtin_amdgcn_wmma_f32_16x16x32_f16(false, a, false, b,
                                                (short)0, c, false, false);
}

// ---- Tensor Data Mover: async 2-D tile load Global -> LDS (ISA 08, D# bitfields) ----
// data_size = 2 bytes; tile = tile_w (contiguous elems) x tile_h rows; row stride in elems.
__device__ __forceinline__ void tdm_load_2d(unsigned lds_byte_off,
                                            unsigned long long gaddr,
                                            unsigned tile_w, unsigned tile_h,
                                            unsigned stride_elems) {
  u32x4 g0;
  g0[0] = 1u;                                          // count=1, user descriptor
  g0[1] = lds_byte_off;                                // LDS destination (bytes)
  g0[2] = (unsigned)(gaddr & 0xffffffffull);           // global_addr[31:0]
  g0[3] = (unsigned)((gaddr >> 32) & 0x1ffffffull) | (2u << 30);  // addr[56:32] | type=2
  const unsigned td0 = stride_elems;                   // tensor_dim0
  const unsigned td1 = 0x100000u;                      // tensor_dim1 (ample, no OOB)
  i32x8 g1;
  g1[0] = (int)(1u << 16);                             // data_size=1 (2 bytes/elem)
  g1[1] = (int)((td0 & 0xffffu) << 16);                // tensor_dim0[15:0] @ bits 63:48
  g1[2] = (int)(((td0 >> 16) & 0xffffu) | ((td1 & 0xffffu) << 16));
  g1[3] = (int)(((td1 >> 16) & 0xffffu) | (tile_w << 16));   // tile_dim0
  g1[4] = (int)(tile_h & 0xffffu);                     // tile_dim1 (tile_dim2 = 0)
  g1[5] = (int)td0;                                    // tensor_dim0_stride[31:0]
  g1[6] = 0;                                           // stride0[47:32], stride1 lo
  g1[7] = 0;
  i32x4 z4;
#pragma unroll
  for (int i = 0; i < 4; ++i) z4[i] = 0;
#if __clang_major__ >= 23
  i32x8 z8;
#pragma unroll
  for (int i = 0; i < 8; ++i) z8[i] = 0;
  __builtin_amdgcn_tensor_load_to_lds(g0, g1, z4, z4, z8, 0);
#else
  __builtin_amdgcn_tensor_load_to_lds(g0, g1, z4, z4, 0);
#endif
}

__device__ __forceinline__ unsigned long long rfl64(unsigned long long v) {
  unsigned lo = __builtin_amdgcn_readfirstlane((unsigned)v);
  unsigned hi = __builtin_amdgcn_readfirstlane((unsigned)(v >> 32));
  return ((unsigned long long)hi << 32) | lo;
}

// TDM-staged, double-buffered GEMM inner loop: one wave computes 16x64.
// Ag: global A pre-offset to (m0, 0); Bg: global B pre-offset to row n0 (= blockIdx.y*512 + wave*64).
// s0/s1: this wave's LDS double buffers, 2560 halves each (B slice 2048 + A 512).
__device__ __forceinline__ void wave_gemm16x64_tdm(const _Float16* Ag,
                                                   const _Float16* Bg,
                                                   int K, int lane,
                                                   _Float16* s0, _Float16* s1,
                                                   v8f acc[4]) {
  const unsigned ldsOff0 = __builtin_amdgcn_readfirstlane((unsigned)(uintptr_t)s0);
  const unsigned ldsOff1 = __builtin_amdgcn_readfirstlane((unsigned)(uintptr_t)s1);
  const unsigned long long aB = rfl64((unsigned long long)(uintptr_t)Ag);
  const unsigned long long bB = rfl64((unsigned long long)(uintptr_t)Bg);
  const int NC = K / 32;

  // prologue: chunk 0 -> buffer 0
  tdm_load_2d(ldsOff0,          bB, 32, 64, (unsigned)K);   // B slice 64x32
  tdm_load_2d(ldsOff0 + 4096u,  aB, 32, 16, (unsigned)K);   // A tile  16x32

  for (int c = 0; c < NC; ++c) {
    if (c + 1 < NC) {
      const unsigned nb = ((c + 1) & 1) ? ldsOff1 : ldsOff0;
      const unsigned long long koff = (unsigned long long)(c + 1) * 64ull; // 32 elems * 2B
      asm volatile("" ::: "memory");
      tdm_load_2d(nb,         bB + koff, 32, 64, (unsigned)K);
      tdm_load_2d(nb + 4096u, aB + koff, 32, 16, (unsigned)K);
      __builtin_amdgcn_s_wait_tensorcnt(2);   // chunk c landed; c+1 in flight
    } else {
      __builtin_amdgcn_s_wait_tensorcnt(0);
    }
    asm volatile("" ::: "memory");
    const _Float16* sb = (c & 1) ? s1 : s0;
    const _Float16* sa = sb + 2048;
    v16h a = load_a_frag(sa, 32, lane);
#pragma unroll
    for (int j = 0; j < 4; ++j) {
      v16h b = load_b_frag(sb + (size_t)(j * 16) * 32, 32, lane);
      acc[j] = wmma16(a, b, acc[j]);
    }
  }
}

// ------------------------- prep kernels -------------------------
__global__ void cvt_f16_kernel(const float* __restrict__ src,
                               _Float16* __restrict__ dst, int n) {
  int i = blockIdx.x * blockDim.x + threadIdx.x;
  if (i < n) dst[i] = (_Float16)src[i];
}

// W[K][N] f32 -> Wt[N][K] f16
__global__ void transpose_cvt_kernel(const float* __restrict__ W,
                                     _Float16* __restrict__ Wt,
                                     int K, int N) {
  int idx = blockIdx.x * blockDim.x + threadIdx.x;
  if (idx >= K * N) return;
  int k = idx / N, n = idx - k * N;
  Wt[(size_t)n * K + k] = (_Float16)W[idx];
}

// ------------------- projection GEMMs with fused PoPE -------------------
#define GEMM_PREAMBLE(BT_PTR)                                                   \
  __shared__ _Float16 smem[2][8][2560];                                         \
  const int lane = threadIdx.x & 31, wave = threadIdx.x >> 5;                   \
  const int nl = lane & 15, g = lane >> 4;                                      \
  const int m0 = blockIdx.x * 16;                                               \
  const int n0 = blockIdx.y * 512 + wave * 64;                                  \
  v8f acc[4];                                                                   \
  _Pragma("unroll") for (int j = 0; j < 4; ++j) acc[j] = zero8();               \
  wave_gemm16x64_tdm(Xh + (size_t)m0 * D_MODEL,                                 \
                     (BT_PTR) + (size_t)n0 * D_MODEL,                           \
                     D_MODEL, lane, &smem[0][wave][0], &smem[1][wave][0], acc);

// Q = Xh(4096x2048) * WqT -> softplus+phase -> Qc (B,H,T,256) f16
__global__ __launch_bounds__(256) void gemm_q_pope(const _Float16* __restrict__ Xh,
                                                   const _Float16* __restrict__ WqT,
                                                   _Float16* __restrict__ Qc) {
  GEMM_PREAMBLE(WqT)
  const float LOGB = 9.210340371976184f;   // ln(10000)
#pragma unroll
  for (int j = 0; j < 4; ++j) {
    const int col = n0 + j * 16 + nl;
    const int hh = col >> 7, d = col & (HD - 1);
    const float freq = __expf(-LOGB * (float)d * (1.0f / (float)HD));
#pragma unroll
    for (int r = 0; r < 8; ++r) {
      const int bt = m0 + 8 * g + r;
      const int bb = bt >> 11, t = bt & (SEQ - 1);
      const float mag = softplusf(acc[j][r]);
      float sn, cs;
      __sincosf((float)t * freq, &sn, &cs);
      const size_t base = (((size_t)(bb * N_HEADS + hh) * SEQ) + t) * QKDIM + d;
      Qc[base]      = (_Float16)(mag * cs);
      Qc[base + HD] = (_Float16)(mag * sn);
    }
  }
}

// K proj: delta is per *query* head -> each kv column feeds 2 heads of Kc.
__global__ __launch_bounds__(256) void gemm_k_pope(const _Float16* __restrict__ Xh,
                                                   const _Float16* __restrict__ WkT,
                                                   const float* __restrict__ delta,
                                                   _Float16* __restrict__ Kc) {
  GEMM_PREAMBLE(WkT)
  const float LOGB = 9.210340371976184f;
  const float NEG2PI = -6.283185307179586f;
#pragma unroll
  for (int j = 0; j < 4; ++j) {
    const int col = n0 + j * 16 + nl;
    const int hk = col >> 7, d = col & (HD - 1);
    const float freq = __expf(-LOGB * (float)d * (1.0f / (float)HD));
    const float dl0 = fminf(0.0f, fmaxf(delta[(hk * 2 + 0) * HD + d], NEG2PI));
    const float dl1 = fminf(0.0f, fmaxf(delta[(hk * 2 + 1) * HD + d], NEG2PI));
#pragma unroll
    for (int r = 0; r < 8; ++r) {
      const int bt = m0 + 8 * g + r;
      const int bb = bt >> 11, t = bt & (SEQ - 1);
      const float mag = softplusf(acc[j][r]);
      const float ang = (float)t * freq;
      float sn, cs;
      __sincosf(ang + dl0, &sn, &cs);
      size_t o0 = (((size_t)(bb * N_HEADS + hk * 2) * SEQ) + t) * QKDIM + d;
      Kc[o0]      = (_Float16)(mag * cs);
      Kc[o0 + HD] = (_Float16)(mag * sn);
      __sincosf(ang + dl1, &sn, &cs);
      size_t o1 = (((size_t)(bb * N_HEADS + hk * 2 + 1) * SEQ) + t) * QKDIM + d;
      Kc[o1]      = (_Float16)(mag * cs);
      Kc[o1 + HD] = (_Float16)(mag * sn);
    }
  }
}

// V proj, written transposed: Vt (B, Hkv, 128, T) f16 so PV B-frags are contiguous.
__global__ __launch_bounds__(256) void gemm_v_t(const _Float16* __restrict__ Xh,
                                                const _Float16* __restrict__ WvT,
                                                _Float16* __restrict__ Vt) {
  GEMM_PREAMBLE(WvT)
#pragma unroll
  for (int j = 0; j < 4; ++j) {
    const int col = n0 + j * 16 + nl;
    const int hk = col >> 7, d = col & (HD - 1);
#pragma unroll
    for (int r = 0; r < 8; ++r) {
      const int bt = m0 + 8 * g + r;
      const int bb = bt >> 11, t = bt & (SEQ - 1);
      Vt[(((size_t)(bb * N_KV + hk) * HD) + d) * SEQ + t] = (_Float16)acc[j][r];
    }
  }
}

// ------------------------- flash attention -------------------------
// Block = 4 waves; each wave owns one 16-row query strip. Causal, online softmax.
__global__ void attn_kernel(const _Float16* __restrict__ Qc,
                            const _Float16* __restrict__ Kc,
                            const _Float16* __restrict__ Vt,
                            _Float16* __restrict__ Yh) {
  __shared__ _Float16 pbuf[4][16][48];   // per-wave P tile (16x32, padded rows)
  const int lane = threadIdx.x & 31, wave = threadIdx.x >> 5;
  const int nl = lane & 15, g = lane >> 4;
  const int bb = blockIdx.z, h = blockIdx.y;
  const int q0 = blockIdx.x * 64 + wave * 16;
  const int hk = h >> 1;                 // GQA: 2 query heads per kv head

  const _Float16* Qb = Qc + (((size_t)(bb * N_HEADS + h) * SEQ) + q0) * QKDIM;
  const _Float16* Kb = Kc + ((size_t)(bb * N_HEADS + h) * SEQ) * QKDIM;
  const _Float16* Vb = Vt + ((size_t)(bb * N_KV + hk) * HD) * SEQ;

  v16h qf[8];
#pragma unroll
  for (int c = 0; c < 8; ++c) qf[c] = load_a_frag(Qb + c * 32, QKDIM, lane);

  v8f o[8];
#pragma unroll
  for (int j = 0; j < 8; ++j) o[j] = zero8();
  float mrow[8], lrow[8];
#pragma unroll
  for (int r = 0; r < 8; ++r) { mrow[r] = -1e30f; lrow[r] = 0.0f; }

  const float scale = 0.08838834764831845f;   // 1/sqrt(128)

  for (int kb = 0; kb <= q0 + 15; kb += 32) {
    __builtin_prefetch(Kb + (size_t)(kb + 32) * QKDIM, 0, 0);
    // S = Q * K^T for 32 keys (two 16-col tiles)
    v8f s1 = zero8(), s2 = zero8();
#pragma unroll
    for (int c = 0; c < 8; ++c) {
      v16h k1 = load_b_frag(Kb + (size_t)kb * QKDIM + c * 32, QKDIM, lane);
      s1 = wmma16(qf[c], k1, s1);
      v16h k2 = load_b_frag(Kb + (size_t)(kb + 16) * QKDIM + c * 32, QKDIM, lane);
      s2 = wmma16(qf[c], k2, s2);
    }
    // causal mask + scale (lane holds key column kb+nl / kb+16+nl; rows 8g+r)
    const int rbase = q0 + 8 * g;
#pragma unroll
    for (int r = 0; r < 8; ++r) {
      const int row = rbase + r;
      s1[r] = (kb + nl      <= row) ? s1[r] * scale : -1e30f;
      s2[r] = (kb + 16 + nl <= row) ? s2[r] * scale : -1e30f;
    }
    // online softmax row statistics (16-lane reductions)
    float sc[8];
#pragma unroll
    for (int r = 0; r < 8; ++r) {
      float mx = fmaxf(s1[r], s2[r]);
#pragma unroll
      for (int off = 1; off < 16; off <<= 1)
        mx = fmaxf(mx, __shfl_xor(mx, off, 32));
      const float mn = fmaxf(mrow[r], mx);
      const float scl = __expf(mrow[r] - mn);
      const float e1 = __expf(s1[r] - mn);
      const float e2 = __expf(s2[r] - mn);
      float rs = e1 + e2;
#pragma unroll
      for (int off = 1; off < 16; off <<= 1)
        rs += __shfl_xor(rs, off, 32);
      lrow[r] = lrow[r] * scl + rs;
      mrow[r] = mn;
      sc[r] = scl;
      s1[r] = e1;
      s2[r] = e2;
    }
#pragma unroll
    for (int j = 0; j < 8; ++j)
#pragma unroll
      for (int r = 0; r < 8; ++r) o[j][r] = o[j][r] * sc[r];

    // transpose P (C-layout -> A-layout) through LDS
#pragma unroll
    for (int r = 0; r < 8; ++r) {
      pbuf[wave][8 * g + r][nl]      = (_Float16)s1[r];
      pbuf[wave][8 * g + r][16 + nl] = (_Float16)s2[r];
    }
    __builtin_amdgcn_wave_barrier();
    asm volatile("s_wait_dscnt 0" ::: "memory");
    v16h pf;
    {
      const _Float16* prow = &pbuf[wave][nl][0];
      union { v16h v; v8h hh[2]; } u;
      u.hh[0] = *(const v8h*)(prow + 8 * g);
      u.hh[1] = *(const v8h*)(prow + 16 + 8 * g);
      pf = u.v;
    }
    __builtin_amdgcn_wave_barrier();
    // O += P * V  (V as Vt rows = dims, contiguous over keys)
#pragma unroll
    for (int j = 0; j < 8; ++j) {
      v16h vf = load_b_frag(Vb + (size_t)(j * 16) * SEQ + kb, SEQ, lane);
      o[j] = wmma16(pf, vf, o[j]);
    }
  }

  // normalize + write Yh (B*T, 2048) f16
#pragma unroll
  for (int r = 0; r < 8; ++r) {
    const int row = q0 + 8 * g + r;
    const float inv = 1.0f / lrow[r];
    const size_t base = ((size_t)(bb * SEQ) + row) * (size_t)D_MODEL + h * HD;
#pragma unroll
    for (int j = 0; j < 8; ++j)
      Yh[base + j * 16 + nl] = (_Float16)(o[j][r] * inv);
  }
}

// ------------------------- output projection -------------------------
__global__ __launch_bounds__(256) void gemm_out(const _Float16* __restrict__ Xh,
                                                const _Float16* __restrict__ WoT,
                                                float* __restrict__ out) {
  GEMM_PREAMBLE(WoT)
#pragma unroll
  for (int j = 0; j < 4; ++j) {
    const int col = n0 + j * 16 + nl;
#pragma unroll
    for (int r = 0; r < 8; ++r)
      out[(size_t)(m0 + 8 * g + r) * D_MODEL + col] = acc[j][r];
  }
}

// ------------------------- launcher -------------------------
extern "C" void kernel_launch(void* const* d_in, const int* in_sizes, int n_in,
                              void* d_out, int out_size, void* d_ws, size_t ws_size,
                              hipStream_t stream) {
  const float* x     = (const float*)d_in[0];
  const float* Wq    = (const float*)d_in[1];
  const float* Wk    = (const float*)d_in[2];
  const float* Wv    = (const float*)d_in[3];
  const float* Wo    = (const float*)d_in[4];
  const float* delta = (const float*)d_in[5];

  char* ws = (char*)d_ws;
  size_t off = 0;
  _Float16* Xh  = (_Float16*)(ws + off); off += (size_t)BT * D_MODEL * 2;        // 16 MB
  _Float16* WqT = (_Float16*)(ws + off); off += (size_t)D_MODEL * D_MODEL * 2;   //  8 MB
  _Float16* WkT = (_Float16*)(ws + off); off += (size_t)1024 * D_MODEL * 2;      //  4 MB
  _Float16* WvT = (_Float16*)(ws + off); off += (size_t)1024 * D_MODEL * 2;      //  4 MB
  _Float16* WoT = (_Float16*)(ws + off); off += (size_t)D_MODEL * D_MODEL * 2;   //  8 MB
  _Float16* Qc  = (_Float16*)(ws + off); off += (size_t)BATCH * N_HEADS * SEQ * QKDIM * 2; // 32 MB
  _Float16* Kc  = (_Float16*)(ws + off); off += (size_t)BATCH * N_HEADS * SEQ * QKDIM * 2; // 32 MB
  _Float16* Vt  = (_Float16*)(ws + off); off += (size_t)BATCH * N_KV * HD * SEQ * 2;       //  8 MB
  _Float16* Yh  = (_Float16*)(ws + off); off += (size_t)BT * D_MODEL * 2;        // 16 MB

  // prep: f32 -> f16 conversions / weight transposes
  {
    int n = BT * D_MODEL;
    cvt_f16_kernel<<<(n + 255) / 256, 256, 0, stream>>>(x, Xh, n);
  }
  transpose_cvt_kernel<<<(D_MODEL * 2048 + 255) / 256, 256, 0, stream>>>(Wq, WqT, D_MODEL, 2048);
  transpose_cvt_kernel<<<(D_MODEL * 1024 + 255) / 256, 256, 0, stream>>>(Wk, WkT, D_MODEL, 1024);
  transpose_cvt_kernel<<<(D_MODEL * 1024 + 255) / 256, 256, 0, stream>>>(Wv, WvT, D_MODEL, 1024);
  transpose_cvt_kernel<<<(2048 * D_MODEL + 255) / 256, 256, 0, stream>>>(Wo, WoT, 2048, D_MODEL);

  // projections with fused PoPE (TDM-staged, double-buffered WMMA GEMMs)
  gemm_q_pope<<<dim3(BT / 16, 2048 / 512), 256, 0, stream>>>(Xh, WqT, Qc);
  gemm_k_pope<<<dim3(BT / 16, 1024 / 512), 256, 0, stream>>>(Xh, WkT, delta, Kc);
  gemm_v_t  <<<dim3(BT / 16, 1024 / 512), 256, 0, stream>>>(Xh, WvT, Vt);

  // flash attention: grid (T/64, H, B), 4 waves per block
  attn_kernel<<<dim3(SEQ / 64, N_HEADS, BATCH), 128, 0, stream>>>(Qc, Kc, Vt, Yh);

  // output projection -> f32
  gemm_out<<<dim3(BT / 16, D_MODEL / 512), 256, 0, stream>>>(Yh, WoT, (float*)d_out);
}
